// LaneSetSuppressor_78700980732218
// MI455X (gfx1250) — 8-bit matrix operand order repaired and checked by reading, compiled, not run on hardware
//
#include <hip/hip_runtime.h>

#define LANES 128
#define NPTS  512
#define CLF_THRESH 0.6f
#define SUP_THRESH 0.05f

typedef int v8i __attribute__((ext_vector_type(8)));

__global__ void lane_prep_kernel(const float* __restrict__ clf,
                                 unsigned char* __restrict__ maskB,
                                 float* __restrict__ conf,
                                 float* __restrict__ out) {
    __shared__ float ssum[256];
    __shared__ int   scnt[256];
    const int l = blockIdx.x;
    const int t = threadIdx.x;
    const float* row = clf + (size_t)l * NPTS;

    float s = 0.0f;
    int   c = 0;
    for (int p = t; p < NPTS; p += 256) {
        float v = row[p];
        unsigned char m = (v > CLF_THRESH) ? 1 : 0;
        maskB[(size_t)l * NPTS + p] = m;
        if (m) { s += v; c++; }
    }
    ssum[t] = s;
    scnt[t] = c;
    __syncthreads();
    for (int off = 128; off > 0; off >>= 1) {
        if (t < off) { ssum[t] += ssum[t + off]; scnt[t] += scnt[t + off]; }
        __syncthreads();
    }
    if (t == 0) {
        conf[l] = ssum[0] / (float)scnt[0];
        out[l]  = 0.0f;
    }
}

__global__ void lane_suppress_kernel(const float* __restrict__ reg,
                                     const unsigned char* __restrict__ maskB,
                                     const float* __restrict__ conf,
                                     float* __restrict__ out) {
    __shared__ unsigned char sMaskI[16 * NPTS];
    __shared__ unsigned char sMaskJ[16 * NPTS];
    __shared__ int sCnt[16 * 16];

    const int tile = blockIdx.x;
    const int ti = (tile >> 3) * 16;
    const int tj = (tile & 7) * 16;
    const int t  = threadIdx.x;

    if (t < 16)      __builtin_prefetch(reg + (size_t)(ti + t) * NPTS, 0, 1);
    else if (t < 32) __builtin_prefetch(reg + (size_t)(tj + (t - 16)) * NPTS, 0, 1);

    const int* gI = (const int*)(maskB + (size_t)ti * NPTS);
    const int* gJ = (const int*)(maskB + (size_t)tj * NPTS);
    int* sI = (int*)sMaskI;
    int* sJ = (int*)sMaskJ;
    for (int k = t; k < 16 * NPTS / 4; k += 256) {
        sI[k] = gI[k];
        sJ[k] = gJ[k];
    }
    __syncthreads();

    if (t < 32) {
        const int lane = t;
        const int hi   = lane >> 4;
        const int mrow = lane & 15;

        v8i acc = {};
        #pragma unroll
        for (int c = 0; c < 8; ++c) {
            v8i a, b;
            #pragma unroll
            for (int v = 0; v < 8; ++v) {
                int ka = ((v & 4) ? 32 : 0) + ((v & 2) ? 16 : 0) +
                         ((v & 1) ? 4 : 0) + hi * 8;
                int kb = ((v & 4) ? 32 : 0) + ((v & 2) ? 16 : 0) +
                         ((v & 1) ? 4 : 0) + hi * 8;
                a[v] = sI[(mrow * NPTS + c * 64 + ka) >> 2];
                b[v] = sJ[(mrow * NPTS + c * 64 + kb) >> 2];
            }
            acc = __builtin_amdgcn_wmma_i32_16x16x64_iu8(
                      false, a, false, b, acc, false, false);
        }
        #pragma unroll
        for (int r = 0; r < 8; ++r)
            sCnt[(r + hi * 8) * 16 + mrow] = acc[r];
    }
    __syncthreads();

    const int M = t >> 4, N = t & 15;
    const int i = ti + M, j = tj + N;
    const float* regI = reg + (size_t)i * NPTS;
    const float* regJ = reg + (size_t)j * NPTS;
    const unsigned char* mi = sMaskI + M * NPTS;
    const unsigned char* mj = sMaskJ + N * NPTS;

    float s = 0.0f;
    #pragma unroll 4
    for (int p = 0; p < NPTS; ++p) {
        if (mi[p] & mj[p]) s += fabsf(regI[p] - regJ[p]);
    }

    const int cnt = sCnt[M * 16 + N];
    const bool flag = (i != j) && (cnt > 0) && (s / (float)cnt < SUP_THRESH);
    if (flag) {
        const float ci = conf[i];
        const float cj = conf[j];
        if (cj > ci) atomicExch(&out[i], 1.0f);
        else         atomicExch(&out[j], 1.0f);
    }
}

extern "C" void kernel_launch(void* const* d_in, const int* in_sizes, int n_in,
                              void* d_out, int out_size, void* d_ws, size_t ws_size,
                              hipStream_t stream) {
    const float* reg_all = (const float*)d_in[0];
    const float* clf_all = (const float*)d_in[1];

    const int total = in_sizes[0];
    const int base  = total - LANES * NPTS;
    const float* reg = reg_all + base;
    const float* clf = clf_all + base;

    unsigned char* maskB = (unsigned char*)d_ws;
    float* conf = (float*)((char*)d_ws + (size_t)LANES * NPTS);
    float* out  = (float*)d_out;

    lane_prep_kernel<<<LANES, 256, 0, stream>>>(clf, maskB, conf, out);
    lane_suppress_kernel<<<64, 256, 0, stream>>>(reg, maskB, conf, out);
}
